// MOTIF_CORD_41961830482443
// MI455X (gfx1250) — matrix operand order repaired and checked by reading, not compiled, not run on hardware
//
#include <hip/hip_runtime.h>

// ---------------- problem constants ----------------
#define P4      4
#define CHN     6
#define DD      6
#define HH      96
#define WW      96
#define KK      1024
#define NN      9216     // H*W
#define CD      24       // CH * 4 effective slices
#define CDP     32       // padded rows for WMMA M-tiles
#define NSTEP_F 288      // N/32  (forward inner steps)
#define KT_F    64       // K/16  (forward col tiles)
#define NT_A    576      // N/16  (adjoint col tiles)
#define NSTEP_A 32       // K/32  (adjoint inner steps)
#define NORMF   192.0f   // 2*sqrt(H*W)
#define GAMMA_C 0.1f
#define TAU_C   0.2f
#define EPS_W   0.01f
#define MCOUNT  98304.0f // B*P*CH*4*K
#define ITERS   2

typedef unsigned int  u32x4 __attribute__((ext_vector_type(4)));
typedef __attribute__((ext_vector_type(16))) __bf16 v16bf;
typedef __attribute__((ext_vector_type(8)))  float  v8f;

union FragU { v16bf v; u32x4 q[2]; };

__device__ __forceinline__ unsigned short f2bf(float f) {
    unsigned int u = __float_as_uint(f);
    u += 0x7FFFu + ((u >> 16) & 1u);   // round-to-nearest-even
    return (unsigned short)(u >> 16);
}

// ---------------- E precompute (two WMMA-fragment layouts, bf16) --------------
// Forward layout: B = E^T, tile (p,kt,s): lane col k = kt*16+(L&15),
// inner n = s*32 + (L>=16?16:0) + j.   frag = 512 ushorts; re then im.
__global__ void k_build_Ef(const float* __restrict__ traj, unsigned short* __restrict__ Ef) {
    size_t i = (size_t)blockIdx.x * 256 + threadIdx.x;
    size_t total = (size_t)P4 * KT_F * NSTEP_F * 512;
    if (i >= total) return;
    int j = (int)(i & 15);
    int lane = (int)((i >> 4) & 31);
    size_t t = i >> 9;
    int s  = (int)(t % NSTEP_F); t /= NSTEP_F;
    int kt = (int)(t % KT_F);
    int p  = (int)(t / KT_F);
    int k = kt * 16 + (lane & 15);
    int n = s * 32 + ((lane >> 4) << 4) + j;
    float ty = traj[(p * 2 + 0) * KK + k];
    float tx = traj[(p * 2 + 1) * KK + k];
    float ph = ty * (float)(n / WW - 48) + tx * (float)(n % WW - 48);
    float sn, cs; sincosf(ph, &sn, &cs);
    size_t base = ((((size_t)p * KT_F + kt) * NSTEP_F + s) * 2) * 512 + (size_t)lane * 16 + j;
    Ef[base]       = f2bf(cs);    // Er
    Ef[base + 512] = f2bf(-sn);   // Ei
}

// Adjoint layout: B = E, tile (p,nt,s): lane col n = nt*16+(L&15),
// inner k = s*32 + (L>=16?16:0) + j.
__global__ void k_build_Ea(const float* __restrict__ traj, unsigned short* __restrict__ Ea) {
    size_t i = (size_t)blockIdx.x * 256 + threadIdx.x;
    size_t total = (size_t)P4 * NT_A * NSTEP_A * 512;
    if (i >= total) return;
    int j = (int)(i & 15);
    int lane = (int)((i >> 4) & 31);
    size_t t = i >> 9;
    int s  = (int)(t % NSTEP_A); t /= NSTEP_A;
    int nt = (int)(t % NT_A);
    int p  = (int)(t / NT_A);
    int n = nt * 16 + (lane & 15);
    int k = s * 32 + ((lane >> 4) << 4) + j;
    float ty = traj[(p * 2 + 0) * KK + k];
    float tx = traj[(p * 2 + 1) * KK + k];
    float ph = ty * (float)(n / WW - 48) + tx * (float)(n % WW - 48);
    float sn, cs; sincosf(ph, &sn, &cs);
    size_t base = ((((size_t)p * NT_A + nt) * NSTEP_A + s) * 2) * 512 + (size_t)lane * 16 + j;
    Ea[base]       = f2bf(cs);
    Ea[base + 512] = f2bf(-sn);
}

// ---------------- init x = nan_to_num(init) ----------------
__global__ void k_init_x(const float* __restrict__ xr0, const float* __restrict__ xi0,
                         float* __restrict__ xr, float* __restrict__ xi) {
    int i = blockIdx.x * 256 + threadIdx.x;
    if (i >= DD * NN) return;
    float a = xr0[i], b = xi0[i];
    if (isnan(a)) a = 0.f; else if (isinf(a)) a = a > 0 ? 3.402823466e38f : -3.402823466e38f;
    if (isnan(b)) b = 0.f; else if (isinf(b)) b = b > 0 ? 3.402823466e38f : -3.402823466e38f;
    xr[i] = a; xi[i] = b;
}

// ---------------- warp (trilinear, mode=nearest) + csm multiply -> bf16 A ----
__global__ void k_warp_csm(const float* __restrict__ xr, const float* __restrict__ xi,
                           const float* __restrict__ mvf,
                           const float* __restrict__ csmr, const float* __restrict__ csmi,
                           unsigned short* __restrict__ Xar, unsigned short* __restrict__ Xai) {
    int i = blockIdx.x * 256 + threadIdx.x;
    if (i >= P4 * 4 * NN) return;
    int n  = i % NN;
    int de = (i / NN) % 4;
    int p  = i / (NN * 4);
    int d  = de + 1;
    float pr, pi;
    if (p < 3) {
        int h = n / WW, w = n % WW;
        float fz = mvf[((size_t)(p * 3 + 0) * DD + d) * NN + n];
        float fy = mvf[((size_t)(p * 3 + 1) * DD + d) * NN + n];
        float fx = mvf[((size_t)(p * 3 + 2) * DD + d) * NN + n];
        float z = fminf(fmaxf((float)d + fz, 0.f), (float)(DD - 1));
        float y = fminf(fmaxf((float)h + fy, 0.f), (float)(HH - 1));
        float x = fminf(fmaxf((float)w + fx, 0.f), (float)(WW - 1));
        int z0 = (int)floorf(z), y0 = (int)floorf(y), x0 = (int)floorf(x);
        int z1 = min(z0 + 1, DD - 1), y1 = min(y0 + 1, HH - 1), x1 = min(x0 + 1, WW - 1);
        float wz = z - z0, wy = y - y0, wx = x - x0;
        auto samp = [&](const float* v) -> float {
            float v000 = v[(z0 * HH + y0) * WW + x0], v001 = v[(z0 * HH + y0) * WW + x1];
            float v010 = v[(z0 * HH + y1) * WW + x0], v011 = v[(z0 * HH + y1) * WW + x1];
            float v100 = v[(z1 * HH + y0) * WW + x0], v101 = v[(z1 * HH + y0) * WW + x1];
            float v110 = v[(z1 * HH + y1) * WW + x0], v111 = v[(z1 * HH + y1) * WW + x1];
            float c00 = v000 + wx * (v001 - v000), c01 = v010 + wx * (v011 - v010);
            float c10 = v100 + wx * (v101 - v100), c11 = v110 + wx * (v111 - v110);
            float c0 = c00 + wy * (c01 - c00), c1 = c10 + wy * (c11 - c10);
            return c0 + wz * (c1 - c0);
        };
        pr = samp(xr); pi = samp(xi);
    } else {
        pr = xr[d * NN + n]; pi = xi[d * NN + n];
    }
    for (int c = 0; c < CHN; ++c) {
        float cr = csmr[(c * DD + d) * NN + n];
        float ci = csmi[(c * DD + d) * NN + n];
        float mr = pr * cr - pi * ci;
        float mi = pr * ci + pi * cr;
        size_t o = ((size_t)p * CDP + (c * 4 + de)) * NN + n;
        Xar[o] = f2bf(mr); Xai[o] = f2bf(mi);
    }
}

// ---------------- forward WMMA: ks^T(24xK) = X^T(24xN) * E^T(NxK) ----------
__global__ void k_fwd(const unsigned short* __restrict__ Ef,
                      const unsigned short* __restrict__ Xar,
                      const unsigned short* __restrict__ Xai,
                      float* __restrict__ ksr, float* __restrict__ ksi) {
    int wave = blockIdx.x * 8 + (threadIdx.x >> 5);
    int lane = threadIdx.x & 31;
    int kt = wave % KT_F, p = wave / KT_F;
    int lh = lane >> 4, ll = lane & 15;
    int c0 = lh * 16;
    const unsigned short* eb  = Ef + (((size_t)p * KT_F + kt) * NSTEP_F) * 1024;
    const unsigned short* ar0 = Xar + (size_t)p * CDP * NN;
    const unsigned short* ai0 = Xai + (size_t)p * CDP * NN;
    v8f z8 = {0.f,0.f,0.f,0.f,0.f,0.f,0.f,0.f};
    v8f acc[2][4];
    for (int m = 0; m < 2; ++m) for (int q = 0; q < 4; ++q) acc[m][q] = z8;
    for (int s = 0; s < NSTEP_F; ++s) {
        const unsigned short* fb = eb + (size_t)s * 1024 + (size_t)lane * 16;
        __builtin_prefetch(fb + 4 * 1024, 0, 1);
        FragU er, ei;
        er.q[0] = *(const u32x4*)(fb);       er.q[1] = *(const u32x4*)(fb + 8);
        ei.q[0] = *(const u32x4*)(fb + 512); ei.q[1] = *(const u32x4*)(fb + 520);
#pragma unroll
        for (int mt = 0; mt < 2; ++mt) {
            int row = mt * 16 + ll;
            const unsigned short* pa = ar0 + (size_t)row * NN + s * 32 + c0;
            const unsigned short* pb = ai0 + (size_t)row * NN + s * 32 + c0;
            FragU fr, fi;
            fr.q[0] = *(const u32x4*)pa; fr.q[1] = *(const u32x4*)(pa + 8);
            fi.q[0] = *(const u32x4*)pb; fi.q[1] = *(const u32x4*)(pb + 8);
            acc[mt][0] = __builtin_amdgcn_wmma_f32_16x16x32_bf16(false, fr.v, false, er.v, (short)0, acc[mt][0], false, false);
            acc[mt][1] = __builtin_amdgcn_wmma_f32_16x16x32_bf16(false, fi.v, false, ei.v, (short)0, acc[mt][1], false, false);
            acc[mt][2] = __builtin_amdgcn_wmma_f32_16x16x32_bf16(false, fr.v, false, ei.v, (short)0, acc[mt][2], false, false);
            acc[mt][3] = __builtin_amdgcn_wmma_f32_16x16x32_bf16(false, fi.v, false, er.v, (short)0, acc[mt][3], false, false);
        }
    }
    int k = kt * 16 + ll;
    const float inv = 1.0f / NORMF;
#pragma unroll
    for (int mt = 0; mt < 2; ++mt)
#pragma unroll
        for (int v = 0; v < 8; ++v) {
            int m = mt * 16 + lh * 8 + v;
            if (m < CD) {
                ksr[((size_t)p * CD + m) * KK + k] = (acc[mt][0][v] - acc[mt][1][v]) * inv;
                ksi[((size_t)p * CD + m) * KK + k] = (acc[mt][2][v] + acc[mt][3][v]) * inv;
            }
        }
}

// ---------------- max |ks| reduction ----------------
__global__ void k_absmax(const float* __restrict__ ksr, const float* __restrict__ ksi,
                         unsigned int* __restrict__ amax) {
    __shared__ float red[256];
    float mx = 0.f;
    size_t total = (size_t)P4 * CD * KK;
    for (size_t i = (size_t)blockIdx.x * 256 + threadIdx.x; i < total; i += (size_t)gridDim.x * 256) {
        float r = ksr[i], q = ksi[i];
        mx = fmaxf(mx, sqrtf(r * r + q * q));
    }
    red[threadIdx.x] = mx; __syncthreads();
    for (int st = 128; st > 0; st >>= 1) {
        if (threadIdx.x < st) red[threadIdx.x] = fmaxf(red[threadIdx.x], red[threadIdx.x + st]);
        __syncthreads();
    }
    if (threadIdx.x == 0) atomicMax(amax, __float_as_uint(red[0]));
}

// ---------------- weighted residual -> bf16 A for adjoint ----------------
__global__ void k_resid(const float* __restrict__ ksr, const float* __restrict__ ksi,
                        const float* __restrict__ kdr, const float* __restrict__ kdi,
                        const unsigned int* __restrict__ amax,
                        unsigned short* __restrict__ gAr, unsigned short* __restrict__ gAi) {
    int i = blockIdx.x * 256 + threadIdx.x;
    if (i >= P4 * CD * KK) return;
    int k = i % KK;
    int m = (i / KK) % CD;
    int p = i / (KK * CD);
    int c = m >> 2, de = m & 3;
    float am = __uint_as_float(*amax);
    float r = ksr[i], q = ksi[i];
    float a = sqrtf(r * r + q * q);
    float w = 1.0f / (a / am + EPS_W);
    float sc = w * w / MCOUNT;
    size_t kidx = (((size_t)p * CHN + c) * DD + (de + 1)) * KK + k;
    float gr = sc * (r - kdr[kidx]);
    float gi = sc * (q - kdi[kidx]);
    size_t o = ((size_t)p * CDP + m) * KK + k;
    gAr[o] = f2bf(gr); gAi[o] = f2bf(gi);
}

// ---------------- adjoint WMMA: y(24xN) = conj-weighted G^H(24xK) * E(KxN) ---
__global__ void k_adj(const unsigned short* __restrict__ Ea,
                      const unsigned short* __restrict__ gAr,
                      const unsigned short* __restrict__ gAi,
                      float* __restrict__ yr, float* __restrict__ yi) {
    int wave = blockIdx.x * 8 + (threadIdx.x >> 5);
    int lane = threadIdx.x & 31;
    int nt = wave % NT_A, p = wave / NT_A;
    int lh = lane >> 4, ll = lane & 15;
    int c0 = lh * 16;
    const unsigned short* eb  = Ea + (((size_t)p * NT_A + nt) * NSTEP_A) * 1024;
    const unsigned short* ar0 = gAr + (size_t)p * CDP * KK;
    const unsigned short* ai0 = gAi + (size_t)p * CDP * KK;
    v8f z8 = {0.f,0.f,0.f,0.f,0.f,0.f,0.f,0.f};
    v8f acc[2][4];
    for (int m = 0; m < 2; ++m) for (int q = 0; q < 4; ++q) acc[m][q] = z8;
    for (int s = 0; s < NSTEP_A; ++s) {
        const unsigned short* fb = eb + (size_t)s * 1024 + (size_t)lane * 16;
        __builtin_prefetch(fb + 4 * 1024, 0, 1);
        FragU er, ei;
        er.q[0] = *(const u32x4*)(fb);       er.q[1] = *(const u32x4*)(fb + 8);
        ei.q[0] = *(const u32x4*)(fb + 512); ei.q[1] = *(const u32x4*)(fb + 520);
#pragma unroll
        for (int mt = 0; mt < 2; ++mt) {
            int row = mt * 16 + ll;
            const unsigned short* pa = ar0 + (size_t)row * KK + s * 32 + c0;
            const unsigned short* pb = ai0 + (size_t)row * KK + s * 32 + c0;
            FragU fr, fi;
            fr.q[0] = *(const u32x4*)pa; fr.q[1] = *(const u32x4*)(pa + 8);
            fi.q[0] = *(const u32x4*)pb; fi.q[1] = *(const u32x4*)(pb + 8);
            acc[mt][0] = __builtin_amdgcn_wmma_f32_16x16x32_bf16(false, fr.v, false, er.v, (short)0, acc[mt][0], false, false);
            acc[mt][1] = __builtin_amdgcn_wmma_f32_16x16x32_bf16(false, fi.v, false, ei.v, (short)0, acc[mt][1], false, false);
            acc[mt][2] = __builtin_amdgcn_wmma_f32_16x16x32_bf16(false, fr.v, false, ei.v, (short)0, acc[mt][2], false, false);
            acc[mt][3] = __builtin_amdgcn_wmma_f32_16x16x32_bf16(false, fi.v, false, er.v, (short)0, acc[mt][3], false, false);
        }
    }
    int n = nt * 16 + ll;
    const float inv = 1.0f / NORMF;
#pragma unroll
    for (int mt = 0; mt < 2; ++mt)
#pragma unroll
        for (int v = 0; v < 8; ++v) {
            int m = mt * 16 + lh * 8 + v;
            if (m < CD) {
                // y = conj(E)*g: y_r = gr*Er + gi*Ei ; y_i = gi*Er - gr*Ei
                yr[((size_t)p * CD + m) * NN + n] = (acc[mt][0][v] + acc[mt][1][v]) * inv;
                yi[((size_t)p * CD + m) * NN + n] = (acc[mt][3][v] - acc[mt][2][v]) * inv;
            }
        }
}

// ---------------- csm^H combine; p==3 adds directly into G ----------------
__global__ void k_csm(const float* __restrict__ yr, const float* __restrict__ yi,
                      const float* __restrict__ csmr, const float* __restrict__ csmi,
                      float* __restrict__ gphr, float* __restrict__ gphi,
                      float* __restrict__ Gr, float* __restrict__ Gi) {
    int i = blockIdx.x * 256 + threadIdx.x;
    if (i >= P4 * 4 * NN) return;
    int n  = i % NN;
    int de = (i / NN) % 4;
    int p  = i / (NN * 4);
    int d  = de + 1;
    float vr = 0.f, vi = 0.f;
    for (int c = 0; c < CHN; ++c) {
        int m = c * 4 + de;
        float a = yr[((size_t)p * CD + m) * NN + n];
        float b = yi[((size_t)p * CD + m) * NN + n];
        float cr = csmr[(c * DD + d) * NN + n];
        float ci = csmi[(c * DD + d) * NN + n];
        vr += cr * a + ci * b;     // conj(csm) * y
        vi += cr * b - ci * a;
    }
    if (p < 3) {
        gphr[((size_t)p * 4 + de) * NN + n] = vr;
        gphi[((size_t)p * 4 + de) * NN + n] = vi;
    } else {
        Gr[d * NN + n] += vr;      // unique (d,n) per thread, G pre-zeroed
        Gi[d * NN + n] += vi;
    }
}

// ---------------- adjoint warp: trilinear scatter with atomics ----------------
__global__ void k_scatter(const float* __restrict__ gphr, const float* __restrict__ gphi,
                          const float* __restrict__ mvf,
                          float* __restrict__ Gr, float* __restrict__ Gi) {
    int i = blockIdx.x * 256 + threadIdx.x;
    if (i >= 3 * 4 * NN) return;
    int n  = i % NN;
    int de = (i / NN) % 4;
    int p  = i / (NN * 4);
    int d  = de + 1;
    int h = n / WW, w = n % WW;
    float fz = mvf[((size_t)(p * 3 + 0) * DD + d) * NN + n];
    float fy = mvf[((size_t)(p * 3 + 1) * DD + d) * NN + n];
    float fx = mvf[((size_t)(p * 3 + 2) * DD + d) * NN + n];
    float z = fminf(fmaxf((float)d + fz, 0.f), (float)(DD - 1));
    float y = fminf(fmaxf((float)h + fy, 0.f), (float)(HH - 1));
    float x = fminf(fmaxf((float)w + fx, 0.f), (float)(WW - 1));
    int z0 = (int)floorf(z), y0 = (int)floorf(y), x0 = (int)floorf(x);
    int z1 = min(z0 + 1, DD - 1), y1 = min(y0 + 1, HH - 1), x1 = min(x0 + 1, WW - 1);
    float wz = z - z0, wy = y - y0, wx = x - x0;
    float vr = gphr[((size_t)p * 4 + de) * NN + n];
    float vi = gphi[((size_t)p * 4 + de) * NN + n];
    int   zi[2] = {z0, z1}, yi2[2] = {y0, y1}, xi2[2] = {x0, x1};
    float wzv[2] = {1.f - wz, wz}, wyv[2] = {1.f - wy, wy}, wxv[2] = {1.f - wx, wx};
#pragma unroll
    for (int a = 0; a < 2; ++a)
#pragma unroll
        for (int b = 0; b < 2; ++b)
#pragma unroll
            for (int c = 0; c < 2; ++c) {
                float wt = wzv[a] * wyv[b] * wxv[c];
                int idx = (zi[a] * HH + yi2[b]) * WW + xi2[c];
                atomicAdd(&Gr[idx], vr * wt);
                atomicAdd(&Gi[idx], vi * wt);
            }
}

// ---------------- TV: normalized forward diffs ----------------
__global__ void k_tv(const float* __restrict__ xr, const float* __restrict__ xi,
                     float* __restrict__ tvr, float* __restrict__ tvi) {
    int i = blockIdx.x * 256 + threadIdx.x;
    if (i >= 4 * NN) return;
    int n = i % NN, de = i / NN, d = de + 1;
    int h = n / WW, w = n % WW;
    for (int comp = 0; comp < 2; ++comp) {
        const float* u = comp ? xi : xr;
        float* tv = comp ? tvi : tvr;
        float u0 = u[d * NN + n];
        float dz = (de < 3)     ? u[(d + 1) * NN + n]  - u0 : 0.f;
        float dy = (h < HH - 1) ? u[d * NN + n + WW]   - u0 : 0.f;
        float dx = (w < WW - 1) ? u[d * NN + n + 1]    - u0 : 0.f;
        float mag = sqrtf(dz * dz + dy * dy + dx * dx + 1e-6f);
        tv[(0 * 4 + de) * NN + n] = dz / mag;
        tv[(1 * 4 + de) * NN + n] = dy / mag;
        tv[(2 * 4 + de) * NN + n] = dx / mag;
    }
}

// ---------------- update x ----------------
__global__ void k_update(float* __restrict__ xr, float* __restrict__ xi,
                         const float* __restrict__ Gr, const float* __restrict__ Gi,
                         const float* __restrict__ tvr, const float* __restrict__ tvi,
                         const float* __restrict__ stdp) {
    int i = blockIdx.x * 256 + threadIdx.x;
    if (i >= DD * NN) return;
    int n = i % NN, d = i / NN;
    int h = n / WW, w = n % WW;
    float s = stdp[0];
    float regr = 0.f, regi = 0.f;
    if (d >= 1 && d <= 4) {
        int de = d - 1;
        for (int comp = 0; comp < 2; ++comp) {
            const float* tv = comp ? tvi : tvr;
            float nz  = tv[(0 * 4 + de) * NN + n];
            float nzm = (de > 0)     ? tv[(0 * 4 + de - 1) * NN + n] : 0.f;
            float ny  = tv[(1 * 4 + de) * NN + n];
            float nym = (h > 0)      ? tv[(1 * 4 + de) * NN + n - WW] : 0.f;
            float nx  = tv[(2 * 4 + de) * NN + n];
            float nxm = (w > 0)      ? tv[(2 * 4 + de) * NN + n - 1] : 0.f;
            float div = (nz - nzm) + (ny - nym) + (nx - nxm);
            if (comp) regi = -div; else regr = -div;
        }
    }
    xr[i] -= GAMMA_C * Gr[i] + TAU_C * s * regr;
    xi[i] -= GAMMA_C * Gi[i] + TAU_C * s * regi;
}

// ---------------- pack output (B,D,H,W,2) ----------------
__global__ void k_out(const float* __restrict__ xr, const float* __restrict__ xi,
                      float* __restrict__ out) {
    int i = blockIdx.x * 256 + threadIdx.x;
    if (i >= DD * NN) return;
    out[(size_t)i * 2 + 0] = xr[i];
    out[(size_t)i * 2 + 1] = xi[i];
}

// ================================================================
extern "C" void kernel_launch(void* const* d_in, const int* in_sizes, int n_in,
                              void* d_out, int out_size, void* d_ws, size_t ws_size,
                              hipStream_t stream) {
    const float* kdr  = (const float*)d_in[0];
    const float* kdi  = (const float*)d_in[1];
    const float* traj = (const float*)d_in[2];
    const float* xr0  = (const float*)d_in[3];
    const float* xi0  = (const float*)d_in[4];
    const float* mvf  = (const float*)d_in[5];
    const float* csmr = (const float*)d_in[6];
    const float* csmi = (const float*)d_in[7];
    const float* stdp = (const float*)d_in[8];
    float* out = (float*)d_out;

    // ---- workspace carve (256B aligned) ----
    char* w = (char*)d_ws;
    auto alloc = [&](size_t bytes) -> void* {
        void* r = (void*)w;
        w += (bytes + 255) & ~(size_t)255;
        return r;
    };
    unsigned short* Ef  = (unsigned short*)alloc((size_t)P4 * KT_F * NSTEP_F * 2 * 512 * 2); // 151MB
    unsigned short* Ea  = (unsigned short*)alloc((size_t)P4 * NT_A * NSTEP_A * 2 * 512 * 2); // 151MB
    unsigned short* Xar = (unsigned short*)alloc((size_t)P4 * CDP * NN * 2);
    unsigned short* Xai = (unsigned short*)alloc((size_t)P4 * CDP * NN * 2);
    unsigned short* gAr = (unsigned short*)alloc((size_t)P4 * CDP * KK * 2);
    unsigned short* gAi = (unsigned short*)alloc((size_t)P4 * CDP * KK * 2);
    float* ksr  = (float*)alloc((size_t)P4 * CD * KK * 4);
    float* ksi  = (float*)alloc((size_t)P4 * CD * KK * 4);
    float* yr   = (float*)alloc((size_t)P4 * CD * NN * 4);
    float* yi   = (float*)alloc((size_t)P4 * CD * NN * 4);
    float* gphr = (float*)alloc((size_t)3 * 4 * NN * 4);
    float* gphi = (float*)alloc((size_t)3 * 4 * NN * 4);
    float* Gr   = (float*)alloc((size_t)DD * NN * 4);
    float* Gi   = (float*)alloc((size_t)DD * NN * 4);
    float* xr   = (float*)alloc((size_t)DD * NN * 4);
    float* xi   = (float*)alloc((size_t)DD * NN * 4);
    float* tvr  = (float*)alloc((size_t)3 * 4 * NN * 4);
    float* tvi  = (float*)alloc((size_t)3 * 4 * NN * 4);
    unsigned int* amax = (unsigned int*)alloc(4);

    // ---- once per call ----
    k_build_Ef<<<147456, 256, 0, stream>>>(traj, Ef);
    k_build_Ea<<<147456, 256, 0, stream>>>(traj, Ea);
    hipMemsetAsync(Xar, 0, (size_t)P4 * CDP * NN * 2, stream);  // zero incl. pad rows
    hipMemsetAsync(Xai, 0, (size_t)P4 * CDP * NN * 2, stream);
    hipMemsetAsync(gAr, 0, (size_t)P4 * CDP * KK * 2, stream);
    hipMemsetAsync(gAi, 0, (size_t)P4 * CDP * KK * 2, stream);
    k_init_x<<<216, 256, 0, stream>>>(xr0, xi0, xr, xi);

    for (int it = 0; it < ITERS; ++it) {
        k_warp_csm<<<576, 256, 0, stream>>>(xr, xi, mvf, csmr, csmi, Xar, Xai);
        k_fwd<<<32, 256, 0, stream>>>(Ef, Xar, Xai, ksr, ksi);
        hipMemsetAsync(amax, 0, 4, stream);
        k_absmax<<<96, 256, 0, stream>>>(ksr, ksi, amax);
        k_resid<<<384, 256, 0, stream>>>(ksr, ksi, kdr, kdi, amax, gAr, gAi);
        hipMemsetAsync(Gr, 0, (size_t)DD * NN * 4, stream);
        hipMemsetAsync(Gi, 0, (size_t)DD * NN * 4, stream);
        k_adj<<<288, 256, 0, stream>>>(Ea, gAr, gAi, yr, yi);
        k_csm<<<576, 256, 0, stream>>>(yr, yi, csmr, csmi, gphr, gphi, Gr, Gi);
        k_scatter<<<432, 256, 0, stream>>>(gphr, gphi, mvf, Gr, Gi);
        k_tv<<<144, 256, 0, stream>>>(xr, xi, tvr, tvi);
        k_update<<<216, 256, 0, stream>>>(xr, xi, Gr, Gi, tvr, tvi, stdp);
    }
    k_out<<<216, 256, 0, stream>>>(xr, xi, out);
}
